// EfficientAttentionBlock_27367531610544
// MI455X (gfx1250) — compile-verified, matrix operand order checked by reading, not run on hardware
//
#include <hip/hip_runtime.h>
#include <hip/hip_bf16.h>
#include <math.h>

#define DMODEL 2048
#define NHEAD  16
#define DHEAD  128
#define SEQ    8192
#define FFNDIM 8192
#define LN_EPS 1e-6f

typedef __bf16 bf16;
typedef __attribute__((ext_vector_type(16))) __bf16 v16bf;
typedef __attribute__((ext_vector_type(8)))  float  v8f;

__device__ __forceinline__ float wave_sum(float v) {
  #pragma unroll
  for (int o = 16; o > 0; o >>= 1) v += __shfl_xor(v, o, 32);
  return v;
}

// ---------------- LayerNorm of a single 2048-vector ----------------
__global__ void ln_vec_kernel(const float* __restrict__ x, const float* __restrict__ g,
                              const float* __restrict__ b, float* __restrict__ y) {
  __shared__ float s1[256], s2[256];
  int t = threadIdx.x;
  float s = 0.f, sq = 0.f;
  for (int i = t; i < DMODEL; i += 256) { float v = x[i]; s += v; sq += v * v; }
  s1[t] = s; s2[t] = sq; __syncthreads();
  for (int o = 128; o > 0; o >>= 1) {
    if (t < o) { s1[t] += s1[t + o]; s2[t] += s2[t + o]; }
    __syncthreads();
  }
  float mean = s1[0] / (float)DMODEL;
  float var  = s2[0] / (float)DMODEL - mean * mean;
  float rstd = rsqrtf(var + LN_EPS);
  for (int i = t; i < DMODEL; i += 256) y[i] = (x[i] - mean) * rstd * g[i] + b[i];
}

// ---------------- Generic GEMV: y[j] = act(bias + sum_d x[d]*W[d,j]) (+res) ----
__global__ void gemv_kernel(const float* __restrict__ x, const float* __restrict__ W,
                            const float* __restrict__ bias, const float* __restrict__ res,
                            float* __restrict__ y, int K, int N, int act) {
  int j = blockIdx.x * blockDim.x + threadIdx.x;
  if (j >= N) return;
  float acc = bias ? bias[j] : 0.f;
  for (int d = 0; d < K; ++d) {
    if ((d & 31) == 0 && (d + 128) < K)
      __builtin_prefetch(&W[(size_t)(d + 128) * N + j], 0, 1);
    acc += x[d] * W[(size_t)d * N + j];
  }
  if (act) {
    float u = acc;
    acc = 0.5f * u * (1.f + tanhf(0.7978845608028654f * (u + 0.044715f * u * u * u)));
  }
  if (res) acc += res[j];
  y[j] = acc;
}

// ---------------- r[d,h] = sum_t wk[d, h*128+t] * q[h*128+t] ----------------
__global__ void build_r_kernel(const float* __restrict__ wk, const float* __restrict__ q,
                               float* __restrict__ r) {
  int tid = blockIdx.x * blockDim.x + threadIdx.x;  // 0..32767
  int d = tid >> 4, h = tid & 15;
  const float4* wr = reinterpret_cast<const float4*>(wk + (size_t)d * DMODEL + h * DHEAD);
  const float4* qr = reinterpret_cast<const float4*>(q + h * DHEAD);
  float acc = 0.f;
  #pragma unroll 8
  for (int t = 0; t < DHEAD / 4; ++t) {
    float4 w4 = wr[t], q4 = qr[t];
    acc += w4.x * q4.x + w4.y * q4.y + w4.z * q4.z + w4.w * q4.w;
  }
  r[d * NHEAD + h] = acc;
}

// ---------------- Pack r [2048,16] into bf16 WMMA B-fragment layout ----------
// rb[step][lane][16 halves]; lane<16: N=lane, K = {0..7,16..23}; lane>=16: N=lane-16, K={8..15,24..31}
__global__ void pack_rb_kernel(const float* __restrict__ r, bf16* __restrict__ rb) {
  int step = blockIdx.x;   // 0..63  (K-step of 32)
  int lane = threadIdx.x;  // 0..31
  int n = lane & 15;
  bool hi = lane >= 16;
  bf16* dst = rb + ((size_t)step * 32 + lane) * 16;
  #pragma unroll
  for (int i = 0; i < 16; ++i) {
    int koff = (i < 8) ? (hi ? i + 8 : i) : (hi ? i + 16 : i + 8);
    dst[i] = (bf16)r[(step * 32 + koff) * NHEAD + n];
  }
}

// ---------------- scores: LN(16 tokens) fused with WMMA [16x2048]x[2048x16] --
__global__ void scores_kernel(const float* __restrict__ toks, const float* __restrict__ g1,
                              const float* __restrict__ b1, const bf16* __restrict__ rb,
                              float* __restrict__ mean_rstd, float* __restrict__ scores) {
  __shared__ float sMR[16][2];
  __shared__ float sRed[8 * 256];
  int tid = threadIdx.x;
  int wave = tid >> 5, lane = tid & 31;
  int s0 = blockIdx.x * 16;

  // phase 1: per-token LN stats (2 tokens per wave)
  for (int tk = wave * 2; tk < wave * 2 + 2; ++tk) {
    const float* row = toks + (size_t)(s0 + tk) * DMODEL;
    float s = 0.f, sq = 0.f;
    for (int i = lane; i < DMODEL; i += 32) { float v = row[i]; s += v; sq += v * v; }
    s = wave_sum(s); sq = wave_sum(sq);
    if (lane == 0) {
      float mean = s / (float)DMODEL;
      float var  = sq / (float)DMODEL - mean * mean;
      float rstd = rsqrtf(var + LN_EPS);
      sMR[tk][0] = mean; sMR[tk][1] = rstd;
      mean_rstd[(size_t)(s0 + tk) * 2 + 0] = mean;
      mean_rstd[(size_t)(s0 + tk) * 2 + 1] = rstd;
    }
  }
  __syncthreads();

  // phase 2: each wave owns a K-chunk of 256; 8 x wmma 16x16x32 bf16 (unrolled
  // so the compiler can pipeline loads/converts of step i+1 under wmma i)
  int m = lane & 15;
  bool hi = lane >= 16;
  float mean = sMR[m][0], rstd = sMR[m][1];
  const float* row = toks + (size_t)(s0 + m) * DMODEL;
  v8f c = {};
  #pragma unroll
  for (int ks = 0; ks < 8; ++ks) {
    int kb = wave * 256 + ks * 32;
    v16bf afrag, bfrag;
    #pragma unroll
    for (int i = 0; i < 16; ++i) {
      int koff = (i < 8) ? (hi ? i + 8 : i) : (hi ? i + 16 : i + 8);
      int k = kb + koff;
      afrag[i] = (bf16)((row[k] - mean) * rstd * g1[k] + b1[k]);
    }
    const bf16* bp = rb + ((size_t)(kb >> 5) * 32 + lane) * 16;
    #pragma unroll
    for (int i = 0; i < 16; ++i) bfrag[i] = bp[i];
    c = __builtin_amdgcn_wmma_f32_16x16x32_bf16(false, afrag, false, bfrag,
                                                (short)0, c, false, false);
  }
  #pragma unroll
  for (int v = 0; v < 8; ++v) sRed[wave * 256 + lane * 8 + v] = c[v];
  __syncthreads();

  // phase 3: cross-wave reduce, write scores[head][token]
  {
    int L = tid >> 3, v = tid & 7;
    float acc = 0.f;
    #pragma unroll
    for (int w = 0; w < 8; ++w) acc += sRed[w * 256 + L * 8 + v];
    int n  = L & 15;
    int mm = v + ((L >= 16) ? 8 : 0);
    scores[(size_t)n * SEQ + (s0 + mm)] = acc;
  }
}

// ---------------- softmax over S per head (scale applied here) ---------------
__global__ void softmax_kernel(float* __restrict__ scores) {
  int h = blockIdx.x, t = threadIdx.x;
  __shared__ float sred[256];
  float* row = scores + (size_t)h * SEQ;
  const float scale = 0.08838834764831845f;  // 128^-0.5
  float mx = -1e30f;
  for (int i = t; i < SEQ; i += 256) mx = fmaxf(mx, row[i]);
  sred[t] = mx; __syncthreads();
  for (int o = 128; o > 0; o >>= 1) {
    if (t < o) sred[t] = fmaxf(sred[t], sred[t + o]);
    __syncthreads();
  }
  mx = sred[0]; __syncthreads();
  float sum = 0.f;
  for (int i = t; i < SEQ; i += 256) {
    float e = __expf((row[i] - mx) * scale);
    row[i] = e; sum += e;
  }
  sred[t] = sum; __syncthreads();
  for (int o = 128; o > 0; o >>= 1) {
    if (t < o) sred[t] += sred[t + o];
    __syncthreads();
  }
  float inv = 1.f / sred[0];
  for (int i = t; i < SEQ; i += 256) row[i] *= inv;
}

__global__ void zero_kernel(float* __restrict__ p, int n) {
  int i = blockIdx.x * blockDim.x + threadIdx.x;
  if (i < n) p[i] = 0.f;
}

// ---------------- a[16,2048] = attn[16,8192] @ LN(all_toks)[8192,2048] (WMMA) -
__global__ void attnctx_kernel(const float* __restrict__ attn, const float* __restrict__ toks,
                               const float* __restrict__ mean_rstd, const float* __restrict__ g1,
                               const float* __restrict__ b1, float* __restrict__ a) {
  int nt = blockIdx.x;       // N tile (16 cols)
  int kc = blockIdx.y;       // K chunk (2048 tokens)
  int lane = threadIdx.x;    // single wave32 block
  int nb = nt * 16;
  int n = lane & 15;
  bool hi = lane >= 16;
  float gc = g1[nb + n], bc = b1[nb + n];
  v8f c = {};
  #pragma unroll 4
  for (int kb = kc * 2048; kb < kc * 2048 + 2048; kb += 32) {
    v16bf afrag, bfrag;
    #pragma unroll
    for (int i = 0; i < 16; ++i) {
      int koff = (i < 8) ? (hi ? i + 8 : i) : (hi ? i + 16 : i + 8);
      int k = kb + koff;
      afrag[i] = (bf16)attn[(size_t)n * SEQ + k];   // A row = head index
      float mean = mean_rstd[(size_t)k * 2 + 0];
      float rstd = mean_rstd[(size_t)k * 2 + 1];
      bfrag[i] = (bf16)((toks[(size_t)k * DMODEL + nb + n] - mean) * rstd * gc + bc);
    }
    c = __builtin_amdgcn_wmma_f32_16x16x32_bf16(false, afrag, false, bfrag,
                                                (short)0, c, false, false);
  }
  #pragma unroll
  for (int v = 0; v < 8; ++v) {
    int mm = v + (hi ? 8 : 0);
    atomicAdd(&a[(size_t)mm * DMODEL + nb + n], c[v]);
  }
}

// ---------------- per-head V projection: out[j] = bv[j] + a[h,:]·wv[:,j] -----
__global__ void headproj_kernel(const float* __restrict__ a, const float* __restrict__ wv,
                                const float* __restrict__ bv, float* __restrict__ y) {
  int j = blockIdx.x * blockDim.x + threadIdx.x;  // 0..2047
  int h = j >> 7;
  float acc = bv[j];
  const float* ar = a + (size_t)h * DMODEL;
  for (int dd = 0; dd < DMODEL; ++dd) {
    if ((dd & 31) == 0 && (dd + 128) < DMODEL)
      __builtin_prefetch(&wv[(size_t)(dd + 128) * DMODEL + j], 0, 1);
    acc += ar[dd] * wv[(size_t)dd * DMODEL + j];
  }
  y[j] = acc;
}

extern "C" void kernel_launch(void* const* d_in, const int* in_sizes, int n_in,
                              void* d_out, int out_size, void* d_ws, size_t ws_size,
                              hipStream_t stream) {
  const float* self_tok = (const float*)d_in[0];
  const float* all_toks = (const float*)d_in[1];
  const float* wq  = (const float*)d_in[2];
  const float* bq  = (const float*)d_in[3];
  const float* wk  = (const float*)d_in[4];
  /* bk (d_in[5]) adds a per-head constant to all scores -> cancels in softmax */
  const float* wv  = (const float*)d_in[6];
  const float* bv  = (const float*)d_in[7];
  const float* wo  = (const float*)d_in[8];
  const float* bo  = (const float*)d_in[9];
  const float* g1  = (const float*)d_in[10];
  const float* b1  = (const float*)d_in[11];
  const float* g2  = (const float*)d_in[12];
  const float* b2  = (const float*)d_in[13];
  const float* w_ffn1 = (const float*)d_in[14];
  const float* b_ffn1 = (const float*)d_in[15];
  const float* w_ffn2 = (const float*)d_in[16];
  const float* b_ffn2 = (const float*)d_in[17];
  float* out = (float*)d_out;

  char* wsb = (char*)d_ws;
  float* nself  = (float*)(wsb + 0);                 // 2048 f
  float* q      = (float*)(wsb + 8192);              // 2048 f
  float* r      = (float*)(wsb + 16384);             // 32768 f  (ends 147456)
  bf16*  rb     = (bf16*)(wsb + 147456);             // 32768 bf16 (ends 212992)
  float* mr     = (float*)(wsb + 212992);            // 16384 f  (ends 278528)
  float* scores = (float*)(wsb + 278528);            // 131072 f (ends 802816)
  float* a      = (float*)(wsb + 802816);            // 32768 f  (ends 933888)
  float* oh     = (float*)(wsb + 933888);            // 2048 f
  float* x      = (float*)(wsb + 942080);            // 2048 f
  float* ng     = (float*)(wsb + 950272);            // 2048 f
  float* h1     = (float*)(wsb + 958464);            // 8192 f   (ends 991232)

  // 1) n_self = LN(self_tok)
  ln_vec_kernel<<<1, 256, 0, stream>>>(self_tok, g1, b1, nself);
  // 2) q = n_self @ wq + bq
  gemv_kernel<<<8, 256, 0, stream>>>(nself, wq, bq, nullptr, q, DMODEL, DMODEL, 0);
  // 3) r[:,h] = Wk(head h block) @ q[h]
  build_r_kernel<<<128, 256, 0, stream>>>(wk, q, r);
  // 4) pack r into bf16 B fragments
  pack_rb_kernel<<<64, 32, 0, stream>>>(r, rb);
  // 5) scores = LN(all_toks) @ r  (fused, WMMA)
  scores_kernel<<<SEQ / 16, 256, 0, stream>>>(all_toks, g1, b1, rb, mr, scores);
  // 6) softmax per head (with 1/sqrt(Dh) scale)
  softmax_kernel<<<NHEAD, 256, 0, stream>>>(scores);
  // 7) a = attn @ LN(all_toks)  (WMMA, K-split with atomics)
  zero_kernel<<<128, 256, 0, stream>>>(a, NHEAD * DMODEL);
  attnctx_kernel<<<dim3(DMODEL / 16, 4), 32, 0, stream>>>(scores, all_toks, mr, g1, b1, a);
  // 8) per-head wv projection
  headproj_kernel<<<8, 256, 0, stream>>>(a, wv, bv, oh);
  // 9) x = self_tok + oh @ wo + bo
  gemv_kernel<<<8, 256, 0, stream>>>(oh, wo, bo, self_tok, x, DMODEL, DMODEL, 0);
  // 10) ng = LN2(x)
  ln_vec_kernel<<<1, 256, 0, stream>>>(x, g2, b2, ng);
  // 11) h1 = gelu(ng @ w_ffn1 + b_ffn1)
  gemv_kernel<<<32, 256, 0, stream>>>(ng, w_ffn1, b_ffn1, nullptr, h1, DMODEL, FFNDIM, 1);
  // 12) out = x + h1 @ w_ffn2 + b_ffn2
  gemv_kernel<<<8, 256, 0, stream>>>(h1, w_ffn2, b_ffn2, x, out, FFNDIM, DMODEL, 0);
  (void)in_sizes; (void)n_in; (void)out_size; (void)ws_size;
}
